// EnKF_LSTM_39865886441902
// MI455X (gfx1250) — matrix operand order repaired and checked by reading, not compiled, not run on hardware
//
#include <hip/hip_runtime.h>
#include <hip/hip_bf16.h>
#include <math.h>

// ---------------------------------------------------------------------------
// EnKF-LSTM for MI455X (gfx1250, wave32, WMMA).
// Dominant work: per-member recurrent GEMM h[32x256] @ W_hh[256x1024],
// done with v_wmma_f32_16x16x32_bf16 (bf16 in, f32 accumulate).
// ---------------------------------------------------------------------------

typedef __attribute__((ext_vector_type(16))) __bf16 v16bf;
typedef __attribute__((ext_vector_type(8)))  __bf16 v8bf;
typedef __attribute__((ext_vector_type(8)))  float  v8f;

#define BS   32
#define SEQ  64
#define DIN  256
#define NHID 256
#define NG   1024        // 4*NHID
#define NENS 64
#define MID  32          // ensemble members (h/c pairs)
#define MOBS 64
#define ROWS (SEQ*BS)    // 2048 rows for the xW GEMM

__device__ __forceinline__ v8f wmma_bf16(v16bf a, v16bf b, v8f c) {
    // 8 args: (neg_a, A, neg_b, B, c_mod, C, reuse_a, reuse_b)
    return __builtin_amdgcn_wmma_f32_16x16x32_bf16(false, a, false, b,
                                                   (short)0, c, false, false);
}
__device__ __forceinline__ float sigmoidf(float x) { return 1.0f / (1.0f + __expf(-x)); }

// --------------------------------------------------------------------------
// 1) x (f32, [BS][SEQ][DIN]) -> xb16 (bf16, row r = t*32+b, [r][k])
// --------------------------------------------------------------------------
__global__ void k_convert_x(const float* __restrict__ x, __bf16* __restrict__ xb) {
    int idx = blockIdx.x * 256 + threadIdx.x;       // 0..524287
    int k   = idx & 255;
    int row = idx >> 8;                             // t*32 + b
    int t   = row >> 5;
    int b   = row & 31;
    xb[idx] = (__bf16)x[(b * SEQ + t) * DIN + k];
}

// --------------------------------------------------------------------------
// 2) Pack W [256][1024] f32 into bf16 WMMA-B fragment tiles.
//    Tile (kt: K-block of 32, cb: N-block of 16). Within a tile, lane L owns
//    16 contiguous bf16 (32 bytes): element j = W[32*kt + (L>>4)*16 + j][16*cb + (L&15)]
// --------------------------------------------------------------------------
__global__ void k_pack_w(const float* __restrict__ W, __bf16* __restrict__ Wp, int ncols) {
    int idx  = blockIdx.x * 256 + threadIdx.x;      // 0..256*ncols-1
    int j    = idx & 15;
    int L    = (idx >> 4) & 31;
    int tile = idx >> 9;
    int nct  = ncols >> 4;
    int cb   = tile % nct;
    int kt   = tile / nct;
    int krow = kt * 32 + (L >> 4) * 16 + j;
    int col  = cb * 16 + (L & 15);
    Wp[idx]  = (__bf16)W[krow * ncols + col];
}

// --------------------------------------------------------------------------
// 3) xW = x @ W_ih + b : [2048][1024] f32.  One 16x16 tile per wave.
// --------------------------------------------------------------------------
__global__ void k_xw_gemm(const __bf16* __restrict__ xb, const __bf16* __restrict__ Wp,
                          const float* __restrict__ bias, float* __restrict__ xw) {
    int wave = threadIdx.x >> 5;
    int lane = threadIdx.x & 31;
    int half = lane >> 4;
    int l15  = lane & 15;
    int tile = blockIdx.x * 8 + wave;               // 0..8191
    int ct   = tile & 63;                           // col-tile (N/16)
    int rt   = tile >> 6;                           // row-tile (2048/16)
    v8f acc  = (v8f){0, 0, 0, 0, 0, 0, 0, 0};
    const __bf16* arow = xb + (rt * 16 + l15) * DIN;
#pragma unroll
    for (int kt = 0; kt < 8; ++kt) {
        union { v16bf v; v8bf h[2]; } a;
        a.h[0] = *(const v8bf*)(arow + kt * 32 + half * 16);
        a.h[1] = *(const v8bf*)(arow + kt * 32 + half * 16 + 8);
        v16bf bf = *(const v16bf*)(Wp + (kt * 64 + ct) * 512 + lane * 16);
        acc = wmma_bf16(a.v, bf, acc);
    }
    int   col = ct * 16 + l15;
    float bb  = bias[col];
#pragma unroll
    for (int e = 0; e < 8; ++e) {
        int row = rt * 16 + half * 8 + e;
        xw[row * NG + col] = acc[e] + bb;
    }
}

// --------------------------------------------------------------------------
// 4) LSTM recurrence. One WG (512 thr = 16 waves) per ensemble member.
//    Per step: 16 waves x 8 tiles x 8 k-iters = 1024 WMMA.
//    LDS: h(bf16) 16KB + c(f32) 32KB + gates(f32) 128KB = 176KB (dyn).
// --------------------------------------------------------------------------
__global__ void k_lstm(const float* __restrict__ enkf, const float* __restrict__ xw,
                       const __bf16* __restrict__ WhhP,
                       const float* __restrict__ eps_h, const float* __restrict__ eps_c,
                       const float* __restrict__ qp, const float* __restrict__ ep,
                       float* __restrict__ stateF) {
    extern __shared__ char smem[];
    __bf16* h_s = (__bf16*)smem;                         // 32*256 bf16
    float*  c_s = (float*)(smem + 16384);                // 32*256 f32
    float*  g_s = (float*)(smem + 16384 + 32768);        // 32*1024 f32
    int m    = blockIdx.x;
    int tid  = threadIdx.x;
    int wave = tid >> 5, lane = tid & 31, half = lane >> 4, l15 = lane & 15;
    float nq = fabsf(qp[0]);
    float ne = fabsf(ep[0]);
#pragma unroll
    for (int i = 0; i < 16; ++i) {
        int p = tid + 512 * i; int b = p >> 8; int n = p & 255;
        h_s[b * NHID + n] = (__bf16)enkf[(b * NHID + n) * NENS + m];
        c_s[b * NHID + n] = enkf[(b * NHID + n) * NENS + MID + m];
    }
    __syncthreads();
    for (int t = 0; t < SEQ; ++t) {
        v8f acc[2][4];
#pragma unroll
        for (int r = 0; r < 2; ++r)
#pragma unroll
            for (int j = 0; j < 4; ++j) acc[r][j] = (v8f){0, 0, 0, 0, 0, 0, 0, 0};
#pragma unroll
        for (int kt = 0; kt < 8; ++kt) {
            union { v16bf v; v8bf h[2]; } a0, a1;
            const __bf16* hr0 = h_s + l15 * NHID + kt * 32 + half * 16;
            a0.h[0] = *(const v8bf*)hr0;
            a0.h[1] = *(const v8bf*)(hr0 + 8);
            const __bf16* hr1 = hr0 + 16 * NHID;
            a1.h[0] = *(const v8bf*)hr1;
            a1.h[1] = *(const v8bf*)(hr1 + 8);
#pragma unroll
            for (int j = 0; j < 4; ++j) {
                int cb = wave * 4 + j;
                v16bf bf = *(const v16bf*)(WhhP + (kt * 64 + cb) * 512 + lane * 16);
                acc[0][j] = wmma_bf16(a0.v, bf, acc[0][j]);
                acc[1][j] = wmma_bf16(a1.v, bf, acc[1][j]);
            }
        }
#pragma unroll
        for (int r = 0; r < 2; ++r)
#pragma unroll
            for (int j = 0; j < 4; ++j) {
                int col = (wave * 4 + j) * 16 + l15;
#pragma unroll
                for (int e = 0; e < 8; ++e)
                    g_s[(r * 16 + half * 8 + e) * NG + col] = acc[r][j][e];
            }
        __syncthreads();
#pragma unroll
        for (int i = 0; i < 16; ++i) {
            int p = tid + 512 * i; int b = p >> 8; int n = p & 255;
            const float* xr = xw + (t * BS + b) * NG;
            float gi = g_s[b * NG + n]       + xr[n];
            float gf = g_s[b * NG + 256 + n] + xr[256 + n];
            float gg = g_s[b * NG + 512 + n] + xr[512 + n];
            float go = g_s[b * NG + 768 + n] + xr[768 + n];
            float c2 = sigmoidf(gf) * c_s[b * NHID + n] + sigmoidf(gi) * tanhf(gg);
            float h2 = sigmoidf(go) * tanhf(c2);
            c_s[b * NHID + n] = c2;
            h_s[b * NHID + n] = (__bf16)h2;
            if (t == SEQ - 1) {
                stateF[(b * NHID + n) * NENS + m]       = h2 + nq * eps_h[m * BS * NHID + b * NHID + n];
                stateF[(b * NHID + n) * NENS + MID + m] = c2 + ne * eps_c[m * BS * NHID + b * NHID + n];
            }
        }
        __syncthreads();
    }
}

// --------------------------------------------------------------------------
// 5) hxi[b][m][N] = Wh @ stateF + bh (forecast obs).  WG per batch.
// --------------------------------------------------------------------------
__global__ void k_obs(const float* __restrict__ S, const float* __restrict__ Wh,
                      const float* __restrict__ bh, float* __restrict__ hxi) {
    extern __shared__ char smem[];
    float* Sl = (float*)smem;   // 256*64 f32 = 64KB
    int b = blockIdx.x, tid = threadIdx.x;
    for (int i = 0; i < 64; ++i) Sl[tid + 256 * i] = S[b * 16384 + tid + 256 * i];
    __syncthreads();
    for (int i = 0; i < 16; ++i) {
        int idx = tid + 256 * i; int mm = idx >> 6; int c = idx & 63;
        float s = bh[mm];
        const float* wr = Wh + mm * NHID;
        for (int n = 0; n < NHID; ++n) s += wr[n] * Sl[n * 64 + c];
        hxi[b * 4096 + idx] = s;
    }
}

// --------------------------------------------------------------------------
// 6) Row means of stateF over ensemble dim.
// --------------------------------------------------------------------------
__global__ void k_mean(const float* __restrict__ S, float* __restrict__ meanF) {
    int b = blockIdx.x, n = threadIdx.x;
    const float* r = S + (b * NHID + n) * NENS;
    float s = 0;
    for (int c = 0; c < NENS; ++c) s += r[c];
    meanF[b * NHID + n] = s * (1.0f / NENS);
}

// --------------------------------------------------------------------------
// 7) Per-batch: HA, P=HA HA^T/63 + r^2 I, Cholesky, solve P X = innov,
//    T[N'][N] = sum_m HA[m][N'] * X[m][N].   64 threads per WG.
// --------------------------------------------------------------------------
__global__ void k_enkf_solve(const float* __restrict__ hxi, const float* __restrict__ y,
                             const float* __restrict__ rp, float* __restrict__ T) {
    extern __shared__ char smem[];
    float* HX  = (float*)smem;   // 4096
    float* HA  = HX + 4096;      // 4096
    float* P   = HA + 4096;      // 4096
    float* SOL = P + 4096;       // 4096
    int b = blockIdx.x, tid = threadIdx.x;   // 64 threads
    float r2 = rp[0] * rp[0];
    for (int i = 0; i < 64; ++i) HX[i * 64 + tid] = hxi[b * 4096 + i * 64 + tid];
    __syncthreads();
    float mu = 0;
    for (int c = 0; c < 64; ++c) mu += HX[tid * 64 + c];
    mu *= (1.0f / 64.0f);
    for (int c = 0; c < 64; ++c) HA[tid * 64 + c] = HX[tid * 64 + c] - mu;
    __syncthreads();
    for (int j = 0; j < 64; ++j) {
        float s = 0;
        for (int c = 0; c < 64; ++c) s += HA[tid * 64 + c] * HA[j * 64 + c];
        P[tid * 64 + j] = s * (1.0f / 63.0f) + ((j == tid) ? r2 : 0.0f);
    }
    __syncthreads();
    for (int k = 0; k < 64; ++k) {                 // Cholesky (lower, in place)
        if (tid == k) P[k * 64 + k] = sqrtf(P[k * 64 + k]);
        __syncthreads();
        if (tid > k) P[tid * 64 + k] /= P[k * 64 + k];
        __syncthreads();
        if (tid > k) {
            float lik = P[tid * 64 + k];
            for (int j = k + 1; j <= tid; ++j) P[tid * 64 + j] -= lik * P[j * 64 + k];
        }
        __syncthreads();
    }
    for (int i = 0; i < 64; ++i) {                 // forward solve, RHS col = tid
        float s = y[b * 64 + i] - HX[i * 64 + tid];
        for (int k = 0; k < i; ++k) s -= P[i * 64 + k] * SOL[k * 64 + tid];
        SOL[i * 64 + tid] = s / P[i * 64 + i];
    }
    for (int i = 63; i >= 0; --i) {                // back solve (L^T)
        float s = SOL[i * 64 + tid];
        for (int k = i + 1; k < 64; ++k) s -= P[k * 64 + i] * SOL[k * 64 + tid];
        SOL[i * 64 + tid] = s / P[i * 64 + i];
    }
    __syncthreads();
    for (int N = 0; N < 64; ++N) {
        float s = 0;
        for (int mm = 0; mm < 64; ++mm) s += HA[mm * 64 + tid] * SOL[mm * 64 + N];
        T[b * 4096 + tid * 64 + N] = s;
    }
}

// --------------------------------------------------------------------------
// 8) stateU = stateF + A @ T / 63  (written straight into d_out state slot).
// --------------------------------------------------------------------------
__global__ void k_state_update(const float* __restrict__ SF, const float* __restrict__ meanF,
                               const float* __restrict__ T, float* __restrict__ outS) {
    extern __shared__ char smem[];
    float* Tl = (float*)smem;    // 4096
    int b  = blockIdx.x >> 6;
    int n0 = (blockIdx.x & 63) * 4;
    int tid = threadIdx.x;
    for (int i = 0; i < 16; ++i) Tl[tid + 256 * i] = T[b * 4096 + tid + 256 * i];
    __syncthreads();
    int col = tid & 63;
    int n   = n0 + (tid >> 6);
    float mean = meanF[b * NHID + n];
    const float* ar = SF + (b * NHID + n) * NENS;
    float s = 0;
    for (int k = 0; k < 64; ++k) s += (ar[k] - mean) * Tl[k * 64 + col];
    int idx = (b * NHID + n) * NENS + col;
    outS[idx] = ar[col] + s * (1.0f / 63.0f);
}

// --------------------------------------------------------------------------
// 9) mu / sig / quad / logdet on (train ? filtered : forecast) ensemble.
// --------------------------------------------------------------------------
__global__ void k_stats(const float* __restrict__ SF, const float* __restrict__ SU,
                        const int* __restrict__ train,
                        const float* __restrict__ Wh, const float* __restrict__ bh,
                        const float* __restrict__ y, const float* __restrict__ rp,
                        float* __restrict__ muOut, float* __restrict__ sigOut,
                        float* __restrict__ quadP, float* __restrict__ logdetP) {
    extern __shared__ char smem[];
    float* Sl  = (float*)smem;     // 16384
    float* HX  = Sl + 16384;       // 4096
    float* P   = HX + 4096;        // 4096
    float* muL = P + 4096;         // 64
    float* zL  = muL + 64;         // 64
    int b = blockIdx.x, tid = threadIdx.x;
    const float* src = (train[0] != 0) ? SU : SF;
    float r2 = rp[0] * rp[0];
    for (int i = 0; i < 64; ++i) Sl[tid + 256 * i] = src[b * 16384 + tid + 256 * i];
    __syncthreads();
    for (int i = 0; i < 16; ++i) {
        int idx = tid + 256 * i; int mm = idx >> 6; int c = idx & 63;
        float s = bh[mm];
        const float* wr = Wh + mm * NHID;
        for (int n = 0; n < NHID; ++n) s += wr[n] * Sl[n * 64 + c];
        HX[idx] = s;
    }
    __syncthreads();
    if (tid < 64) {
        float mu = 0;
        for (int c = 0; c < 64; ++c) mu += HX[tid * 64 + c];
        mu *= (1.0f / 64.0f);
        muL[tid] = mu;
        muOut[b * 64 + tid] = mu;
    }
    __syncthreads();
    for (int i = 0; i < 16; ++i) {
        int idx = tid + 256 * i;
        HX[idx] -= muL[idx >> 6];
    }
    __syncthreads();
    for (int i = 0; i < 16; ++i) {
        int idx = tid + 256 * i; int ii = idx >> 6; int jj = idx & 63;
        float s = 0;
        for (int c = 0; c < 64; ++c) s += HX[ii * 64 + c] * HX[jj * 64 + c];
        s = s * (1.0f / 63.0f) + ((ii == jj) ? r2 : 0.0f);
        P[idx] = s;
        sigOut[b * 4096 + idx] = s;
    }
    __syncthreads();
    if (tid < 64) P[tid * 64 + tid] += 1e-6f;
    __syncthreads();
    for (int k = 0; k < 64; ++k) {
        if (tid == k) P[k * 64 + k] = sqrtf(P[k * 64 + k]);
        __syncthreads();
        if (tid > k && tid < 64) P[tid * 64 + k] /= P[k * 64 + k];
        __syncthreads();
        if (tid > k && tid < 64) {
            float lik = P[tid * 64 + k];
            for (int j = k + 1; j <= tid; ++j) P[tid * 64 + j] -= lik * P[j * 64 + k];
        }
        __syncthreads();
    }
    if (tid == 0) {
        float quad = 0, logdet = 0;
        for (int i = 0; i < 64; ++i) {
            float s = y[b * 64 + i] - muL[i];
            for (int k = 0; k < i; ++k) s -= P[i * 64 + k] * zL[k];
            zL[i] = s / P[i * 64 + i];
            logdet += __logf(P[i * 64 + i]);
        }
        for (int i = 63; i >= 0; --i) {
            float s = zL[i];
            for (int k = i + 1; k < 64; ++k) s -= P[k * 64 + i] * zL[k];
            zL[i] = s / P[i * 64 + i];
        }
        for (int i = 0; i < 64; ++i) quad += (y[b * 64 + i] - muL[i]) * zL[i];
        quadP[b]   = quad;
        logdetP[b] = 2.0f * logdet;
    }
}

__global__ void k_reduce(const float* __restrict__ quadP, const float* __restrict__ logdetP,
                         float* __restrict__ out) {
    if (threadIdx.x == 0) {
        float l = 0, nis = 0;
        for (int b = 0; b < 32; ++b) {
            l   += -0.5f * logdetP[b] - 0.5f * quadP[b];
            nis += quadP[b];
        }
        out[0] = l;
        out[1] = nis * (1.0f / 32.0f);
    }
}

// --------------------------------------------------------------------------
extern "C" void kernel_launch(void* const* d_in, const int* in_sizes, int n_in,
                              void* d_out, int out_size, void* d_ws, size_t ws_size,
                              hipStream_t stream) {
    const float* x     = (const float*)d_in[0];
    const float* y     = (const float*)d_in[1];
    const float* enkf  = (const float*)d_in[2];
    const float* W_ih  = (const float*)d_in[3];
    const float* W_hh  = (const float*)d_in[4];
    const float* bvec  = (const float*)d_in[5];
    const float* Wh    = (const float*)d_in[6];
    const float* bh    = (const float*)d_in[7];
    const float* q     = (const float*)d_in[8];
    const float* e     = (const float*)d_in[9];
    const float* r     = (const float*)d_in[10];
    const float* eps_h = (const float*)d_in[11];
    const float* eps_c = (const float*)d_in[12];
    const int*   train = (const int*)d_in[13];

    char* ws = (char*)d_ws;
    __bf16* xb16    = (__bf16*)(ws + 0);            // 1,048,576 B
    __bf16* WihP    = (__bf16*)(ws + 1048576);      //   524,288 B
    __bf16* WhhP    = (__bf16*)(ws + 1572864);      //   524,288 B
    float*  xW      = (float*)(ws + 2097152);       // 8,388,608 B
    float*  stateF  = (float*)(ws + 10485760);      // 2,097,152 B
    float*  hxiF    = (float*)(ws + 12582912);      //   524,288 B
    float*  meanF   = (float*)(ws + 13107200);      //    32,768 B
    float*  Tm      = (float*)(ws + 13139968);      //   524,288 B
    float*  quadP   = (float*)(ws + 13664256);      //       128 B
    float*  logdetP = (float*)(ws + 13664384);      //       128 B

    float* muOut  = (float*)d_out;            // [32][64]
    float* sigOut = (float*)d_out + 2048;     // [32][64][64]
    float* stOut  = (float*)d_out + 133120;   // [1][32][256][64]
    float* scOut  = (float*)d_out + 657408;   // l, nis

    k_convert_x<<<2048, 256, 0, stream>>>(x, xb16);
    k_pack_w<<<1024, 256, 0, stream>>>(W_ih, WihP, NG);
    k_pack_w<<<1024, 256, 0, stream>>>(W_hh, WhhP, NG);
    k_xw_gemm<<<1024, 256, 0, stream>>>(xb16, WihP, bvec, xW);
    k_lstm<<<32, 512, 16384 + 32768 + 131072, stream>>>(enkf, xW, WhhP, eps_h, eps_c,
                                                        q, e, stateF);
    k_obs<<<32, 256, 65536, stream>>>(stateF, Wh, bh, hxiF);
    k_mean<<<32, 256, 0, stream>>>(stateF, meanF);
    k_enkf_solve<<<32, 64, 65536, stream>>>(hxiF, y, r, Tm);
    k_state_update<<<2048, 256, 16384, stream>>>(stateF, meanF, Tm, stOut);
    k_stats<<<32, 256, (16384 + 4096 + 4096 + 64 + 64) * 4, stream>>>(
        stateF, stOut, train, Wh, bh, y, r, muOut, sigOut, quadP, logdetP);
    k_reduce<<<1, 32, 0, stream>>>(quadP, logdetP, scOut);
}
